// CrossAttention_28252294873566
// MI455X (gfx1250) — matrix operand order repaired and checked by reading, not compiled, not run on hardware
//
#include <hip/hip_runtime.h>

#define HEADS 8
#define DHEAD 64
#define NKV   512
#define DIM   256
#define INNER 512
#define TOPK  4
#define NB    2      // batches per block (amortizes weight reads from L2)

typedef __attribute__((ext_vector_type(16))) __bf16 v16bf;
typedef __attribute__((ext_vector_type(8)))  float  v8f;

union V16U {
    uint4 q[2];
    v16bf v;
    unsigned short us[16];
};

static __device__ inline unsigned short f2bf(float f) {
    union { float f; unsigned u; } x; x.f = f;
    unsigned r = x.u + 0x7FFFu + ((x.u >> 16) & 1u);   // round-to-nearest-even
    return (unsigned short)(r >> 16);
}

// ---- LDS arena layout (bytes); total 325632 <= 327680 (320 KB/WGP) ----
// Aliased regions noted by live phase.
#define OFF_KV   0         // ushort[512*256]          262144  (per-sub-batch kv, bf16)
#define OFF_SCR  262144    // float [4096]  16384      qin(ph1) | scores(ph2) | head_sum | vh(ph3)
#define OFF_ATT  278528    // ushort[8*512]  8192      attn bf16 (WMMA A)
#define OFF_QK   286720    // ushort[NB][256*16] 16384 Wk^T q, WMMA B (cols 8..15 zero)
#define OFF_CTX  303104    // float [NB][8*256] 16384
#define OFF_Q    319488    // float [NB][512]    4096
#define OFF_MASK 323584    // int   [512]        2048  (per-sub-batch)
#define SMEM_BYTES 325632

__global__ __launch_bounds__(256) void ca_kernel(
    const float* __restrict__ inp_q, const float* __restrict__ inp_kv,
    const int* __restrict__ mask,
    const float* __restrict__ Wq, const float* __restrict__ Wk,
    const float* __restrict__ Wv, const float* __restrict__ Wo,
    const float* __restrict__ bo, float* __restrict__ out)
{
    __shared__ __align__(16) char smem[SMEM_BYTES];

    unsigned short* kvLds   = (unsigned short*)(smem + OFF_KV);
    float*          qinLds  = (float*)(smem + OFF_SCR);   // [NB][256], phase 1 only
    float*          scoresLds = (float*)(smem + OFF_SCR); // [8][512], phase 2
    float*          hsLds   = (float*)(smem + OFF_SCR);   // [512], after softmax (reads attn only)
    float*          vhLds   = (float*)(smem + OFF_SCR);   // [NB][512], phase 3
    unsigned short* attnLds = (unsigned short*)(smem + OFF_ATT);
    unsigned short* qkLds   = (unsigned short*)(smem + OFF_QK);
    float*          ctxLds  = (float*)(smem + OFF_CTX);
    float*          qLds    = (float*)(smem + OFF_Q);
    int*            maskLds = (int*)(smem + OFF_MASK);

    const int b0   = blockIdx.x * NB;
    const int tid  = threadIdx.x;
    const int lane = tid & 31;
    const int wave = tid >> 5;
    const int Btot = gridDim.x * NB;

    // Warm L2 for the shared weights (global_prefetch_b8 path).
    __builtin_prefetch(&Wq[tid * 64], 0, 0);
    __builtin_prefetch(&Wk[tid * 64], 0, 0);
    __builtin_prefetch(&Wv[tid * 64], 0, 0);
    __builtin_prefetch(&Wo[tid * 64], 0, 0);

    // ======== phase 1: q projection + qk = scale * Wk^T q, both sub-batches ========
    qinLds[0 * DIM + tid] = inp_q[(size_t)b0 * DIM + tid];
    qinLds[1 * DIM + tid] = inp_q[(size_t)(b0 + 1) * DIM + tid];
    __syncthreads();

    {   // q[nb][i] = inp_q[b0+nb] @ Wq ; Wq read ONCE per block
        float a00 = 0.f, a01 = 0.f, a10 = 0.f, a11 = 0.f;
        for (int d = 0; d < DIM; ++d) {
            float w0 = Wq[(size_t)d * INNER + tid];
            float w1 = Wq[(size_t)d * INNER + tid + 256];
            float q0 = qinLds[d], q1 = qinLds[DIM + d];
            a00 += q0 * w0; a01 += q0 * w1;
            a10 += q1 * w0; a11 += q1 * w1;
        }
        qLds[0 * INNER + tid]       = a00;
        qLds[0 * INNER + tid + 256] = a01;
        qLds[1 * INNER + tid]       = a10;
        qLds[1 * INNER + tid + 256] = a11;
    }
    __syncthreads();

    // qk[nb][h][c] = 0.125 * sum_d q[nb][h*64+d] * Wk[c, h*64+d] ; Wk read ONCE
    for (int p = tid; p < HEADS * DIM; p += 256) {
        int h = p >> 8;          // p / 256
        int c = p & 255;
        const float* wkp = Wk + (size_t)c * INNER + h * DHEAD;
        const float* q0p = qLds + h * DHEAD;
        const float* q1p = qLds + INNER + h * DHEAD;
        float a0 = 0.f, a1 = 0.f;
        for (int d = 0; d < DHEAD; ++d) {
            float w = wkp[d];
            a0 += q0p[d] * w;
            a1 += q1p[d] * w;
        }
        qkLds[0 * DIM * 16 + c * 16 + h] = f2bf(a0 * 0.125f);
        qkLds[1 * DIM * 16 + c * 16 + h] = f2bf(a1 * 0.125f);
    }
    for (int c = tid; c < DIM; c += 256)
        for (int n = HEADS; n < 16; ++n) {
            qkLds[0 * DIM * 16 + c * 16 + n] = 0;
            qkLds[1 * DIM * 16 + c * 16 + n] = 0;
        }
    __syncthreads();

    // ======== phase 2: per sub-batch attention (stage kv once, 2x WMMA pass) ========
    for (int nb = 0; nb < NB; ++nb) {
        const int b = b0 + nb;

        // ---- stage inp_kv[b] -> LDS bf16 (the HBM-dominant stream) + mask ----
        maskLds[tid]       = mask[(size_t)b * NKV + tid];
        maskLds[tid + 256] = mask[(size_t)b * NKV + tid + 256];
        {
            const float4* src = (const float4*)(inp_kv + (size_t)b * NKV * DIM);
            unsigned int* dst = (unsigned int*)kvLds;
            #pragma unroll 8
            for (int i = 0; i < (NKV * DIM / 4) / 256; ++i) {   // 128 iterations
                int g = tid + 256 * i;
                float4 f = src[g];
                unsigned lo = ((unsigned)f2bf(f.y) << 16) | f2bf(f.x);
                unsigned hi = ((unsigned)f2bf(f.w) << 16) | f2bf(f.z);
                dst[2 * g]     = lo;
                dst[2 * g + 1] = hi;
            }
        }
        __syncthreads();

        // ---- scores[j,h] = kv[j,:] . qk[:,h] via v_wmma_f32_16x16x32_bf16 ----
        {
            const int row  = lane & 15;
            const int kb8  = (lane >> 4) * 8;
            const int kb16 = (lane >> 4) * 16;
            const int col  = lane & 15;
            const uint4* kv4 = (const uint4*)kvLds;
            const unsigned short* qkp = qkLds + nb * DIM * 16;
            for (int t = wave; t < NKV / 16; t += 8) {   // 32 j-tiles, 4 per wave
                const int j0 = t * 16;
                v8f acc = {};
                #pragma unroll
                for (int cc = 0; cc < DIM / 32; ++cc) {
                    const int c0 = cc * 32;
                    V16U a;
                    int base = ((j0 + row) * DIM + c0 + kb8) >> 3;
                    a.q[0] = kv4[base];
                    a.q[1] = kv4[base + 2];
                    V16U bb;
                    #pragma unroll
                    for (int e = 0; e < 16; ++e)
                        bb.us[e] = qkp[(c0 + kb8 + (e & 7) + ((e >> 3) << 4)) * 16 + col];
                    acc = __builtin_amdgcn_wmma_f32_16x16x32_bf16(
                            false, a.v, false, bb.v, (short)0, acc, false, false);
                }
                if (col < HEADS) {
                    int jb = j0 + ((lane >> 4) * 8);
                    #pragma unroll
                    for (int r = 0; r < 8; ++r)
                        scoresLds[col * NKV + jb + r] = acc[r];
                }
            }
        }
        __syncthreads();

        // ---- masked softmax, one wave per head ----
        {
            const int h = wave;
            float vals[16];
            float m = -3.0e38f;
            #pragma unroll
            for (int i = 0; i < 16; ++i) {
                int j = lane + 32 * i;
                float s = scoresLds[h * NKV + j] + (maskLds[j] ? 0.f : -10000.f);
                vals[i] = s;
                m = fmaxf(m, s);
            }
            for (int off = 16; off > 0; off >>= 1) m = fmaxf(m, __shfl_xor(m, off));
            float sum = 0.f;
            #pragma unroll
            for (int i = 0; i < 16; ++i) { vals[i] = __expf(vals[i] - m); sum += vals[i]; }
            for (int off = 16; off > 0; off >>= 1) sum += __shfl_xor(sum, off);
            float inv = 1.f / sum;
            #pragma unroll
            for (int i = 0; i < 16; ++i) {
                int j = lane + 32 * i;
                attnLds[h * NKV + j] = f2bf(vals[i] * inv);
            }
        }
        __syncthreads();   // scores region now dead -> reused as head_sum

        // ---- head_sum[j] = sum_h attn[h,j] (from bf16; 0.4% err is fine for argmax) ----
        for (int j = tid; j < NKV; j += 256) {
            float s = 0.f;
            #pragma unroll
            for (int h = 0; h < HEADS; ++h) {
                union { unsigned u; float f; } x;
                x.u = (unsigned)attnLds[h * NKV + j] << 16;
                s += x.f;
            }
            hsLds[j] = s;
        }
        __syncthreads();

        // ---- top-4 (wave 0): sequential argmax, lowest-index tie-break ----
        if (tid < 32) {
            float lv[16];
            #pragma unroll
            for (int i = 0; i < 16; ++i) lv[i] = hsLds[lane + 32 * i];
            for (int t = 0; t < TOPK; ++t) {
                float bv = -3.0e38f; int bi = NKV;
                #pragma unroll
                for (int i = 0; i < 16; ++i) {
                    int j = lane + 32 * i;
                    if (lv[i] > bv || (lv[i] == bv && j < bi)) { bv = lv[i]; bi = j; }
                }
                for (int off = 16; off > 0; off >>= 1) {
                    float ov = __shfl_xor(bv, off);
                    int   oi = __shfl_xor(bi, off);
                    if (ov > bv || (ov == bv && oi < bi)) { bv = ov; bi = oi; }
                }
                if (lane == 0) out[(size_t)Btot * DIM + b * TOPK + t] = (float)bi;
                if ((bi & 31) == lane) lv[bi >> 5] = -3.0e38f;
            }
        }

        // ---- ctx[h,c] = sum_j attn[h,j] * kv[j,c] via WMMA (M=heads,K=j,N=c) ----
        {
            const int row  = lane & 15;
            const int kb8  = (lane >> 4) * 8;
            const int kb16 = (lane >> 4) * 16;
            const int col  = lane & 15;
            const uint4* at4 = (const uint4*)attnLds;
            float* ctxp = ctxLds + nb * HEADS * DIM;
            for (int t = wave; t < DIM / 16; t += 8) {   // 16 col-tiles, 2 per wave
                const int c0 = t * 16;
                v8f acc = {};
                for (int kk = 0; kk < NKV / 32; ++kk) {
                    const int j0 = kk * 32;
                    V16U a;
                    if (row < HEADS) {
                        int base = (row * NKV + j0 + kb8) >> 3;
                        a.q[0] = at4[base];
                        a.q[1] = at4[base + 2];
                    } else {
                        a.q[0] = make_uint4(0, 0, 0, 0);
                        a.q[1] = make_uint4(0, 0, 0, 0);
                    }
                    V16U bb;
                    #pragma unroll
                    for (int e = 0; e < 16; ++e)
                        bb.us[e] = kvLds[(j0 + kb8 + (e & 7) + ((e >> 3) << 4)) * DIM + c0 + col];
                    acc = __builtin_amdgcn_wmma_f32_16x16x32_bf16(
                            false, a.v, false, bb.v, (short)0, acc, false, false);
                }
                if (lane < 16) {   // lanes >=16 hold padded rows 8..15
                    #pragma unroll
                    for (int r = 0; r < 8; ++r)
                        ctxp[r * DIM + c0 + col] = acc[r];
                }
            }
        }
        __syncthreads();   // protects kv restage (next nb) and scores reuse
    }

    // ======== phase 3: vh[nb][h*64+d] = ctx[nb][h,:] @ Wv[:,h*64+d] ; Wv ONCE ========
    for (int i = tid; i < INNER; i += 256) {
        int h = i >> 6;
        const float* c0p = ctxLds + h * DIM;
        const float* c1p = ctxLds + HEADS * DIM + h * DIM;
        float a0 = 0.f, a1 = 0.f;
        for (int c = 0; c < DIM; ++c) {
            float w = Wv[(size_t)c * INNER + i];
            a0 += c0p[c] * w;
            a1 += c1p[c] * w;
        }
        vhLds[0 * INNER + i] = a0;
        vhLds[1 * INNER + i] = a1;
    }
    __syncthreads();

    // ======== phase 4: out[b,:] = vh @ Wo + bo ; Wo ONCE ========
    {
        float bias = bo[tid];
        float a0 = bias, a1 = bias;
        for (int i = 0; i < INNER; ++i) {
            float w = Wo[(size_t)i * DIM + tid];
            a0 += vhLds[i] * w;
            a1 += vhLds[INNER + i] * w;
        }
        out[(size_t)b0 * DIM + tid]       = a0;
        out[(size_t)(b0 + 1) * DIM + tid] = a1;
    }
}

extern "C" void kernel_launch(void* const* d_in, const int* in_sizes, int n_in,
                              void* d_out, int out_size, void* d_ws, size_t ws_size,
                              hipStream_t stream) {
    const float* inp_q  = (const float*)d_in[0];
    const float* inp_kv = (const float*)d_in[1];
    const int*   mask   = (const int*)d_in[2];
    const float* Wq     = (const float*)d_in[3];
    const float* Wk     = (const float*)d_in[4];
    const float* Wv     = (const float*)d_in[5];
    const float* Wo     = (const float*)d_in[6];
    const float* bo     = (const float*)d_in[7];
    // d_in[8] is topk (==4), fixed at compile time.
    float* out = (float*)d_out;
    const int B = in_sizes[0] / DIM;   // 512 batches
    hipLaunchKernelGGL(ca_kernel, dim3(B / NB), dim3(256), 0, stream,
                       inp_q, inp_kv, mask, Wq, Wk, Wv, Wo, bo, out);
}
